// RWKVGroundedMemory_16999480558027
// MI455X (gfx1250) — matrix operand order repaired and checked by reading, not compiled, not run on hardware
//
#include <hip/hip_runtime.h>
#include <hip/hip_bf16.h>

typedef __bf16 bf16;
typedef __attribute__((ext_vector_type(16))) __bf16 v16bf;
typedef __attribute__((ext_vector_type(8)))  __bf16 v8bf;
typedef __attribute__((ext_vector_type(8)))  float  v8f;
typedef unsigned int u32x4 __attribute__((ext_vector_type(4)));
typedef int          i32x4 __attribute__((ext_vector_type(4)));
typedef int          i32x8 __attribute__((ext_vector_type(8)));

constexpr int Bc = 4, Lc = 2048, Dc = 1024, Hc = 16, HDc = 64;
constexpr int Mrows = Bc * Lc;                 // 8192
constexpr int BM = 128, BN = 64, BK = 32;
constexpr int LDA = BK + 8;                    // padded LDS row (80B)
constexpr int LDB = BK + 8;                    // matches TDM pad: 64B row + 16B pad

// toolchain dispatch for tensor_load_to_lds arity (ROCm7.2: 5 args, therock: 6)
#if defined(__has_include)
# if __has_include(<hip/amd_detail/amd_gfx1250_TDM.h>)
#  define TDM_6ARG 1
# endif
#endif

// ---------------------------------------------------------------------------
// f32 -> bf16 conversion (grid-stride)
// ---------------------------------------------------------------------------
__global__ __launch_bounds__(256) void cvt_f32_bf16(const float* __restrict__ in,
                                                    bf16* __restrict__ out, int n) {
    int i = blockIdx.x * blockDim.x + threadIdx.x;
    int stride = gridDim.x * blockDim.x;
    for (; i < n; i += stride) out[i] = (bf16)in[i];
}

// ---------------------------------------------------------------------------
// CDNA5 async global->LDS copy of 16 bytes per lane (ASYNCcnt).
// ---------------------------------------------------------------------------
__device__ __forceinline__ void async_ld16(unsigned lds_off, const bf16* g) {
    asm volatile("global_load_async_to_lds_b128 %0, %1, off"
                 :: "v"(lds_off), "v"(g) : "memory");
}
__device__ __forceinline__ void wait_async_le2() {
    asm volatile("s_wait_asynccnt 2" ::: "memory");
}
__device__ __forceinline__ void wait_async_le0() {
    asm volatile("s_wait_asynccnt 0" ::: "memory");
}

// ---------------------------------------------------------------------------
// TDM: issue one 2D tile load (BN x BK bf16, tensor row stride Dc) into LDS
// with padding 16B after every 64B row -> LDS row stride = LDB elements.
// D# layout per CDNA5 ISA.
// ---------------------------------------------------------------------------
__device__ __forceinline__ i32x8 tdm_group1() {
    i32x8 g1;
    g1[0] = (1 << 16)        // data_size = 1 -> 2 bytes
          | (1 << 20)        // pad_enable
          | (3 << 22)        // pad_interval: 16 DWORDs (64B) before padding
          | (3 << 25);       // pad_amount: 4 DWORDs (16B)
    g1[1] = (Dc & 0xffff) << 16;                  // tensor_dim0[15:0] (abar=0)
    g1[2] = ((Dc >> 16) & 0xffff) | ((Dc & 0xffff) << 16);  // dim0 hi | dim1 lo
    g1[3] = ((Dc >> 16) & 0xffff) | (BK << 16);   // dim1 hi | tile_dim0 = 32
    g1[4] = BN;                                   // tile_dim1 = 64, tile_dim2 = 0
    g1[5] = Dc;                                   // tensor_dim0_stride low32
    g1[6] = 0;                                    // stride hi | dim1_stride lo
    g1[7] = 0;
    return g1;
}

__device__ __forceinline__ void tdm_issue(unsigned lds_byte, const bf16* gaddr,
                                          i32x8 g1) {
    unsigned long long ga = (unsigned long long)(uintptr_t)gaddr;
    u32x4 g0;
    g0[0] = 1u;                                   // count=1 (valid descriptor)
    g0[1] = lds_byte;                             // lds_addr
    g0[2] = (unsigned)ga;                         // global_addr[31:0]
    g0[3] = (unsigned)(ga >> 32) | (2u << 30);    // global_addr[56:32] | type=2
    i32x4 z4 = {};
#ifdef TDM_6ARG
    i32x8 z8 = {};
    __builtin_amdgcn_tensor_load_to_lds(g0, g1, z4, z4, z8, 0);
#else
    __builtin_amdgcn_tensor_load_to_lds(g0, g1, z4, z4, 0);
#endif
}

// ---------------------------------------------------------------------------
// Fused GEMM: Cm[m,n] = sum_k A[m,k] * Wm[n,k]  for NB weight matrices.
// Block: 256 threads (8 waves), tile BM x BN, K-step BK=32, double-buffered.
//  - A tile:  per-lane global_load_async_to_lds_b128 (ASYNCcnt)
//  - W tiles: one TDM tensor_load_to_lds descriptor each, wave 0 (TENSORcnt)
//  - compute: v_wmma_f32_16x16x32_bf16, NB*4 accumulators per wave
// ---------------------------------------------------------------------------
template <int NB, int OUT_BF16>
__global__ __launch_bounds__(256) void wmma_gemm_nt(
    const bf16* __restrict__ A,
    const bf16* __restrict__ W0, const bf16* __restrict__ W1,
    const bf16* __restrict__ W2,
    void* __restrict__ o0, void* __restrict__ o1, void* __restrict__ o2) {
    __shared__ bf16 As[2][BM * LDA];
    __shared__ bf16 Bs[2][NB][BN * LDB];

    const bf16* Ws[3]  = {W0, W1, W2};
    void*       outs[3] = {o0, o1, o2};

    const int tid  = threadIdx.x;
    const int wave = tid >> 5;
    const int lane = tid & 31;

    const int rowBase = blockIdx.x * BM;
    const int colBase = blockIdx.y * BN;

    // A-tile staging coordinates (2 x 16 bytes per thread)
    const int ar = tid >> 1;             // 0..127 : A tile row
    const int ak = (tid & 1) << 4;       // 0 / 16 : A tile k offset (16 elems)
    const bf16* aptr = A + (size_t)(rowBase + ar) * Dc + ak;
    const unsigned aOff = (unsigned)(uintptr_t)&As[0][ar * LDA + ak];
    const unsigned aStr = (unsigned)(BM * LDA * sizeof(bf16));

    const i32x8 g1 = tdm_group1();
    const size_t wTile = (size_t)colBase * Dc;   // element offset of tile row 0

    v8f acc[NB][4] = {};

    // fragment-gather coordinates (contiguous k order k = 16h + s, same for A and B)
    const int arow = wave * 16 + (lane & 15);
    const int akb  = (lane < 16) ? 0 : 16;      // A: lanes<16 hold K 0..15
    const int bkb  = (lane < 16) ? 0 : 16;      // B: lanes<16 hold K 0..15

    constexpr int nK = Dc / BK;                  // 32 K-steps

    // prime the pipeline: tile 0 into buffer 0
    async_ld16(aOff,      aptr);
    async_ld16(aOff + 16, aptr + 8);
    if (wave == 0) {
#pragma unroll
        for (int m = 0; m < NB; ++m)
            tdm_issue((unsigned)(uintptr_t)&Bs[0][m][0], Ws[m] + wTile, g1);
    }

    for (int kt = 0; kt < nK; ++kt) {
        const int cur = kt & 1;

        if (kt + 1 < nK) {
            const int nxt = (kt + 1) & 1;
            const bf16* aN = aptr + (size_t)(kt + 1) * BK;
            __builtin_prefetch(aN + BK, 0, 1);     // global_prefetch_b8 (+2 tiles)
            async_ld16(aOff + nxt * aStr,      aN);
            async_ld16(aOff + nxt * aStr + 16, aN + 8);
            if (wave == 0) {
#pragma unroll
                for (int m = 0; m < NB; ++m)
                    tdm_issue((unsigned)(uintptr_t)&Bs[nxt][m][0],
                              Ws[m] + wTile + (size_t)(kt + 1) * BK, g1);
                __builtin_amdgcn_s_wait_tensorcnt(NB);  // tile kt's TDM done
            }
            wait_async_le2();                           // tile kt's A done
        } else {
            if (wave == 0) __builtin_amdgcn_s_wait_tensorcnt(0);
            wait_async_le0();
        }
        __syncthreads();        // publish staged tile to all waves

        // A fragment: two 16B chunks per lane
        v8bf alo = *(const v8bf*)&As[cur][arow * LDA + akb];
        v8bf ahi = *(const v8bf*)&As[cur][arow * LDA + akb + 8];
        v16bf afrag;
#pragma unroll
        for (int i = 0; i < 8; ++i) { afrag[i] = alo[i]; afrag[8 + i] = ahi[i]; }

#pragma unroll
        for (int m = 0; m < NB; ++m) {
#pragma unroll
            for (int j = 0; j < 4; ++j) {
                const int bcol = j * 16 + (lane & 15);
                v8bf blo = *(const v8bf*)&Bs[cur][m][bcol * LDB + bkb];
                v8bf bhi = *(const v8bf*)&Bs[cur][m][bcol * LDB + bkb + 8];
                v16bf bfrag;
#pragma unroll
                for (int i = 0; i < 8; ++i) { bfrag[i] = blo[i]; bfrag[8 + i] = bhi[i]; }
                acc[m][j] = __builtin_amdgcn_wmma_f32_16x16x32_bf16(
                    false, afrag, false, bfrag, (short)0, acc[m][j], false, false);
            }
        }
        __syncthreads();        // all waves done with buf[cur] before reuse
    }

    // epilogue: C/D layout -> VGPR i : M = i (+8 for lanes>=16), N = lane&15
    const int ro = (lane < 16) ? 0 : 8;
    const int cl = lane & 15;
#pragma unroll
    for (int m = 0; m < NB; ++m) {
#pragma unroll
        for (int j = 0; j < 4; ++j) {
#pragma unroll
            for (int i = 0; i < 8; ++i) {
                const size_t grow = (size_t)(rowBase + wave * 16 + ro + i);
                const size_t gcol = (size_t)(colBase + j * 16 + cl);
                if (OUT_BF16)
                    ((bf16*)outs[m])[grow * Dc + gcol] = (bf16)acc[m][j][i];
                else
                    ((float*)outs[m])[grow * Dc + gcol] = acc[m][j][i];
            }
        }
    }
}

// ---------------------------------------------------------------------------
// Sequential RWKV scan: one 64-thread block per (b,h).
// Thread e owns state column state[d][e], d=0..63, in registers.
// ---------------------------------------------------------------------------
__global__ __launch_bounds__(64) void rwkv_scan(const bf16* __restrict__ kb,
                                                const bf16* __restrict__ vb,
                                                const bf16* __restrict__ rb,
                                                const float* __restrict__ td,
                                                const float* __restrict__ tf,
                                                bf16* __restrict__ yb,
                                                float* __restrict__ stateOut) {
    const int b = blockIdx.x >> 4;   // / H
    const int h = blockIdx.x & 15;   // % H
    const int e = threadIdx.x;       // 0..63

    __shared__ float sk[HDc], sr[HDc], sdec[HDc], stf_s[HDc];
    sdec[e]  = td[h * HDc + e];
    stf_s[e] = tf[h * HDc + e];

    float st[HDc];
#pragma unroll
    for (int d = 0; d < HDc; ++d) st[d] = 0.0f;
    __syncthreads();

    size_t base = ((size_t)b * Lc) * Dc + (size_t)h * HDc;
    for (int t = 0; t < Lc; ++t, base += Dc) {
        float kv = (float)kb[base + e];
        float rp = (float)rb[base + e];
        float ve = (float)vb[base + e];
        sk[e] = kv;
        sr[e] = 1.0f / (1.0f + __expf(-rp));
        __syncthreads();

        float y = 0.0f;
#pragma unroll
        for (int d = 0; d < HDc; ++d) {
            st[d] = st[d] * sdec[d] + sk[d] * ve * stf_s[d];
            y += st[d] * sr[d];
        }
        yb[base + e] = (bf16)y;
        __syncthreads();
    }

#pragma unroll
    for (int d = 0; d < HDc; ++d)
        stateOut[(((size_t)b * Hc + h) * HDc + d) * HDc + e] = st[d];
}

// ---------------------------------------------------------------------------
extern "C" void kernel_launch(void* const* d_in, const int* in_sizes, int n_in,
                              void* d_out, int out_size, void* d_ws, size_t ws_size,
                              hipStream_t stream) {
    const float* x  = (const float*)d_in[0];
    const float* Wk = (const float*)d_in[1];
    const float* Wv = (const float*)d_in[2];
    const float* Wr = (const float*)d_in[3];
    const float* Wo = (const float*)d_in[4];
    const float* td = (const float*)d_in[5];
    const float* tf = (const float*)d_in[6];

    const size_t nX = (size_t)Mrows * Dc;   // 8,388,608
    const size_t nW = (size_t)Dc * Dc;      // 1,048,576

    char* w = (char*)d_ws;
    bf16* xb   = (bf16*)w; w += nX * sizeof(bf16);
    bf16* Wkb  = (bf16*)w; w += nW * sizeof(bf16);
    bf16* Wvb  = (bf16*)w; w += nW * sizeof(bf16);
    bf16* Wrb  = (bf16*)w; w += nW * sizeof(bf16);
    bf16* Wob  = (bf16*)w; w += nW * sizeof(bf16);
    bf16* kbuf = (bf16*)w; w += nX * sizeof(bf16);
    bf16* vbuf = (bf16*)w; w += nX * sizeof(bf16);
    bf16* rbuf = (bf16*)w; w += nX * sizeof(bf16);
    bf16* ybuf = (bf16*)w; w += nX * sizeof(bf16);

    float* yOut     = (float*)d_out;
    float* stateOut = (float*)d_out + nX;

    cvt_f32_bf16<<<4096, 256, 0, stream>>>(x,  xb,  (int)nX);
    cvt_f32_bf16<<<1024, 256, 0, stream>>>(Wk, Wkb, (int)nW);
    cvt_f32_bf16<<<1024, 256, 0, stream>>>(Wv, Wvb, (int)nW);
    cvt_f32_bf16<<<1024, 256, 0, stream>>>(Wr, Wrb, (int)nW);
    cvt_f32_bf16<<<1024, 256, 0, stream>>>(Wo, Wob, (int)nW);

    dim3 g(Mrows / BM, Dc / BN);   // 64 x 16 blocks

    // fused k,v,r projection: one pass over x, 3 TDM weight streams
    wmma_gemm_nt<3, 1><<<g, 256, 0, stream>>>(xb, Wkb, Wvb, Wrb,
                                              kbuf, vbuf, rbuf);

    rwkv_scan<<<Bc * Hc, HDc, 0, stream>>>(kbuf, vbuf, rbuf, td, tf, ybuf, stateOut);

    // output projection: f32 result straight into d_out
    wmma_gemm_nt<1, 0><<<g, 256, 0, stream>>>(ybuf, Wob, nullptr, nullptr,
                                              yOut, nullptr, nullptr);
}
